// MultiModalCrossAttention_6906307412019
// MI455X (gfx1250) — 16-bit and 8-bit matrix operand order repaired and checked by reading, compiled, not run on hardware
//
#include <hip/hip_runtime.h>
#include <stdint.h>


typedef __attribute__((ext_vector_type(16))) _Float16 v16h;
typedef __attribute__((ext_vector_type(8)))  _Float16 v8h;
typedef __attribute__((ext_vector_type(8)))  float    v8f;
typedef __attribute__((ext_vector_type(8)))  int      v8i;

constexpr int   BATCH   = 2;
constexpr int   SEQ     = 2048;
constexpr int   DMODEL  = 1024;
constexpr int   HEADS   = 16;
constexpr int   DHEAD   = 64;
constexpr int   TOKENS  = BATCH * SEQ;
constexpr float ATTSCALE = 0.125f;
constexpr float EPSF    = 1e-5f;

__device__ __forceinline__ v16h cat8(v8h a, v8h b) {
  return __builtin_shufflevector(a, b, 0,1,2,3,4,5,6,7,8,9,10,11,12,13,14,15);
}
__device__ __forceinline__ v8h ldv8h(const _Float16* p) {
  return *reinterpret_cast<const v8h*>(p);
}

__device__ __forceinline__ uint32_t lds_off32(const void* p) {
  return (uint32_t)(uintptr_t)p;
}

__device__ __forceinline__ void async_copy_b128(void* lds_dst, const void* gsrc) {
  asm volatile("global_load_async_to_lds_b128 %0, %1, off"
               :: "v"(lds_off32(lds_dst)), "v"(gsrc)
               : "memory");
}
__device__ __forceinline__ void wait_asynccnt0() {
  asm volatile("s_wait_asynccnt 0x0" ::: "memory");
}
__device__ __forceinline__ void wait_dscnt0() {
  asm volatile("s_wait_dscnt 0x0" ::: "memory");
}

__global__ __launch_bounds__(1024) void weight_quant_kernel(
    const float* __restrict__ W0, const float* __restrict__ W1,
    const float* __restrict__ W2, const float* __restrict__ W3,
    int8_t* __restrict__ Q0, int8_t* __restrict__ Q1,
    int8_t* __restrict__ Q2, int8_t* __restrict__ Q3,
    float* __restrict__ wsOut)
{
  const int mat = blockIdx.x;
  const float* W = (mat == 0) ? W0 : (mat == 1) ? W1 : (mat == 2) ? W2 : W3;
  int8_t*      Q = (mat == 0) ? Q0 : (mat == 1) ? Q1 : (mat == 2) ? Q2 : Q3;
  const int nElem = DMODEL * DMODEL;

  float s = 0.f;
  for (int i = threadIdx.x; i < nElem; i += 1024) s += fabsf(W[i]);
  #pragma unroll
  for (int o = 16; o > 0; o >>= 1) s += __shfl_xor(s, o, 32);

  __shared__ float red[32];
  __shared__ float wsShared;
  if ((threadIdx.x & 31) == 0) red[threadIdx.x >> 5] = s;
  __syncthreads();
  if (threadIdx.x == 0) {
    float t = 0.f;
    #pragma unroll
    for (int i = 0; i < 32; ++i) t += red[i];
    const float mean = t / (float)nElem;
    const float ws = 1.0f / fmaxf(mean, EPSF);
    wsShared = ws;
    wsOut[mat] = ws;
  }
  __syncthreads();
  const float ws = wsShared;
  for (int i = threadIdx.x; i < nElem; i += 1024) {
    float q = rintf(W[i] * ws);
    q = fminf(1.0f, fmaxf(-1.0f, q));
    Q[i] = (int8_t)q;
  }
}

__global__ __launch_bounds__(256) void act_quant_kernel(
    const float* __restrict__ X, int8_t* __restrict__ Xq, float* __restrict__ xs)
{
  const int row = blockIdx.x;
  const int tid = threadIdx.x;
  const float4 v = *reinterpret_cast<const float4*>(X + (size_t)row * DMODEL + tid * 4);
  float amax = fmaxf(fmaxf(fabsf(v.x), fabsf(v.y)), fmaxf(fabsf(v.z), fabsf(v.w)));
  #pragma unroll
  for (int o = 16; o > 0; o >>= 1) amax = fmaxf(amax, __shfl_xor(amax, o, 32));

  __shared__ float red[8];
  if ((tid & 31) == 0) red[tid >> 5] = amax;
  __syncthreads();
  float m = red[0];
  #pragma unroll
  for (int i = 1; i < 8; ++i) m = fmaxf(m, red[i]);

  const float s = 127.0f / fmaxf(m, EPSF);
  if (tid == 0) xs[row] = s;

  const int q0 = (int)fminf(127.f, fmaxf(-128.f, rintf(v.x * s)));
  const int q1 = (int)fminf(127.f, fmaxf(-128.f, rintf(v.y * s)));
  const int q2 = (int)fminf(127.f, fmaxf(-128.f, rintf(v.z * s)));
  const int q3 = (int)fminf(127.f, fmaxf(-128.f, rintf(v.w * s)));
  const int packed = (q0 & 255) | ((q1 & 255) << 8) | ((q2 & 255) << 16) | ((q3 & 255) << 24);
  *reinterpret_cast<int*>(Xq + (size_t)row * DMODEL + tid * 4) = packed;
}

template <typename OutT>
__global__ __launch_bounds__(256) void gemm_iu8_kernel(
    const int8_t* __restrict__ Aq, const int8_t* __restrict__ Wq,
    const float* __restrict__ rowScale, const float* __restrict__ wScalePtr,
    OutT* __restrict__ Out, float extraScale)
{
  constexpr int DK = DMODEL;
  constexpr int DN = DMODEL;
  const int wave = blockIdx.x * 8 + (threadIdx.x >> 5);
  const int lane = threadIdx.x & 31;
  const int lr   = lane & 15;
  const int half = lane >> 4;
  const int rowG = wave / (DN / 64);
  const int colG = wave % (DN / 64);
  const int rowBase = rowG * 32;
  const int colBase = colG * 64;

  const int8_t* arow0 = Aq + (size_t)(rowBase + lr) * DK + half * 16;
  const int8_t* arow1 = arow0 + (size_t)16 * DK;
  const int8_t* brow0 = Wq + (size_t)(colBase + lr) * DK + half * 16;
  const int8_t* brow1 = brow0 + (size_t)16 * DK;
  const int8_t* brow2 = brow0 + (size_t)32 * DK;
  const int8_t* brow3 = brow0 + (size_t)48 * DK;

  v8i acc[2][4];
  const v8i zi = {0, 0, 0, 0, 0, 0, 0, 0};
  #pragma unroll
  for (int i = 0; i < 2; ++i)
    #pragma unroll
    for (int t = 0; t < 4; ++t) acc[i][t] = zi;

  #pragma unroll 2
  for (int k = 0; k < DK; k += 64) {
    __builtin_prefetch(arow0 + k + 256, 0, 1);
    __builtin_prefetch(brow0 + k + 256, 0, 1);
    v8i Am[2], Bm[4];
    {
      const int8_t* ap = arow0 + k;
      #pragma unroll
      for (int i = 0; i < 2; ++i, ap = arow1 + k) {
        const int2 a0 = *reinterpret_cast<const int2*>(ap);
        const int2 a1 = *reinterpret_cast<const int2*>(ap + 8);
        const int2 a2 = *reinterpret_cast<const int2*>(ap + 32);
        const int2 a3 = *reinterpret_cast<const int2*>(ap + 40);
        Am[i] = (v8i){a0.x, a0.y, a1.x, a1.y, a2.x, a2.y, a3.x, a3.y};
      }
      const int8_t* bp[4] = {brow0 + k, brow1 + k, brow2 + k, brow3 + k};
      #pragma unroll
      for (int t = 0; t < 4; ++t) {
        const int4 b0 = *reinterpret_cast<const int4*>(bp[t]);
        const int4 b1 = *reinterpret_cast<const int4*>(bp[t] + 32);
        Bm[t] = (v8i){b0.x, b0.y, b0.z, b0.w, b1.x, b1.y, b1.z, b1.w};
      }
    }
    #pragma unroll
    for (int i = 0; i < 2; ++i)
      #pragma unroll
      for (int t = 0; t < 4; ++t)
        acc[i][t] = __builtin_amdgcn_wmma_i32_16x16x64_iu8(true, Am[i], true, Bm[t],
                                                           acc[i][t], false, false);
  }

  const float wS = wScalePtr[0];
  #pragma unroll
  for (int i = 0; i < 2; ++i) {
    #pragma unroll
    for (int r = 0; r < 8; ++r) {
      const int t = rowBase + i * 16 + r + 8 * half;
      const float sc = extraScale / (rowScale[t] * wS);
      OutT* orow = Out + (size_t)t * DN + colBase + lr;
      #pragma unroll
      for (int tc = 0; tc < 4; ++tc) orow[tc * 16] = (OutT)((float)acc[i][tc][r] * sc);
    }
  }
}

__global__ __launch_bounds__(256) void attention_kernel(
    const _Float16* __restrict__ Qh, const _Float16* __restrict__ Kh,
    const _Float16* __restrict__ Vh, float* __restrict__ Oatt)
{
  constexpr int KC = 64;
  constexpr int PITCH = 72;
  __shared__ __attribute__((aligned(128))) _Float16 KsBuf[2][KC * PITCH];
  __shared__ __attribute__((aligned(128))) _Float16 VtBuf[2][DHEAD * PITCH];
  __shared__ __attribute__((aligned(128))) _Float16 Pbuf[8][16 * PITCH];

  const int tid  = threadIdx.x;
  const int widx = tid >> 5;
  const int lane = tid & 31;
  const int lr   = lane & 15;
  const int half = lane >> 4;

  const int grp = blockIdx.x;
  const int qt  = (grp % (SEQ / 128)) * 8 + widx;
  const int h   = (grp / (SEQ / 128)) % HEADS;
  const int b   = grp / ((SEQ / 128) * HEADS);

  const _Float16* kbase = Kh + (size_t)b * SEQ * DMODEL + h * DHEAD;
  const _Float16* vbase = Vh + (size_t)b * SEQ * DMODEL + h * DHEAD;
  _Float16* pP = &Pbuf[widx][0];

  const _Float16* qrow = Qh + ((size_t)(b * SEQ + qt * 16 + lr)) * DMODEL + h * DHEAD;
  const v16h A0 = cat8(ldv8h(qrow + half * 16), ldv8h(qrow + half * 16 + 8));
  const v16h A1 = cat8(ldv8h(qrow + 32 + half * 16), ldv8h(qrow + 32 + half * 16 + 8));

  float rowM[8], rowL[8];
  v8f Oacc[4];
  const v8f zf = {0, 0, 0, 0, 0, 0, 0, 0};
  #pragma unroll
  for (int r = 0; r < 8; ++r) { rowM[r] = -1e30f; rowL[r] = 0.f; }
  #pragma unroll
  for (int t = 0; t < 4; ++t) Oacc[t] = zf;

  auto stage = [&](int db, int jj) {
    const int row = tid >> 2;
    const int seg = (tid & 3) * 16;
    const _Float16* gk = kbase + (size_t)(jj + row) * DMODEL + seg;
    _Float16* lk = &KsBuf[db][row * PITCH + seg];
    async_copy_b128(lk, gk);
    async_copy_b128(lk + 8, gk + 8);
    const v16h vv = *reinterpret_cast<const v16h*>(vbase + (size_t)(jj + row) * DMODEL + seg);
    #pragma unroll
    for (int e = 0; e < 16; ++e) VtBuf[db][(seg + e) * PITCH + row] = vv[e];
  };

  stage(0, 0);
  wait_asynccnt0();
  __syncthreads();

  for (int j = 0; j < SEQ; j += KC) {
    const int db = (j >> 6) & 1;
    if (j + KC < SEQ) stage(db ^ 1, j + KC);
    const _Float16* Ksc = &KsBuf[db][0];
    const _Float16* Vtc = &VtBuf[db][0];

    v8f S[4];
    #pragma unroll
    for (int s = 0; s < 4; ++s) {
      const _Float16* kr = Ksc + (s * 16 + lr) * PITCH + half * 16;
      const v16h B0 = cat8(ldv8h(kr), ldv8h(kr + 8));
      const v16h B1 = cat8(ldv8h(kr + 32), ldv8h(kr + 40));
      v8f c = zf;
      c = __builtin_amdgcn_wmma_f32_16x16x32_f16(false, A0, false, B0, (short)0, c, false, false);
      c = __builtin_amdgcn_wmma_f32_16x16x32_f16(false, A1, false, B1, (short)0, c, false, false);
      S[s] = c;
    }

    #pragma unroll
    for (int r = 0; r < 8; ++r) {
      float mx = fmaxf(fmaxf(S[0][r], S[1][r]), fmaxf(S[2][r], S[3][r]));
      #pragma unroll
      for (int o = 1; o < 16; o <<= 1) mx = fmaxf(mx, __shfl_xor(mx, o, 32));
      const float mnew = fmaxf(rowM[r], mx);
      const float corr = __expf(rowM[r] - mnew);
      rowM[r] = mnew;
      float p[4], ps = 0.f;
      #pragma unroll
      for (int s = 0; s < 4; ++s) { p[s] = __expf(S[s][r] - mnew); ps += p[s]; }
      #pragma unroll
      for (int o = 1; o < 16; o <<= 1) ps += __shfl_xor(ps, o, 32);
      rowL[r] = rowL[r] * corr + ps;
      const int m = r + 8 * half;
      #pragma unroll
      for (int s = 0; s < 4; ++s) pP[m * PITCH + s * 16 + lr] = (_Float16)p[s];
      #pragma unroll
      for (int t = 0; t < 4; ++t) Oacc[t][r] *= corr;
    }

    wait_dscnt0();

    const _Float16* pr = pP + lr * PITCH + half * 16;
    const v16h AP0 = cat8(ldv8h(pr), ldv8h(pr + 8));
    const v16h AP1 = cat8(ldv8h(pr + 32), ldv8h(pr + 40));

    #pragma unroll
    for (int t = 0; t < 4; ++t) {
      const _Float16* vr = Vtc + (t * 16 + lr) * PITCH + half * 16;
      const v16h BV0 = cat8(ldv8h(vr), ldv8h(vr + 8));
      const v16h BV1 = cat8(ldv8h(vr + 32), ldv8h(vr + 40));
      Oacc[t] = __builtin_amdgcn_wmma_f32_16x16x32_f16(false, AP0, false, BV0, (short)0,
                                                       Oacc[t], false, false);
      Oacc[t] = __builtin_amdgcn_wmma_f32_16x16x32_f16(false, AP1, false, BV1, (short)0,
                                                       Oacc[t], false, false);
    }

    wait_asynccnt0();
    __syncthreads();
  }

  #pragma unroll
  for (int r = 0; r < 8; ++r) {
    const float inv = 1.0f / rowL[r];
    float* orow = Oatt + ((size_t)(b * SEQ + qt * 16 + r + 8 * half)) * DMODEL + h * DHEAD + lr;
    #pragma unroll
    for (int t = 0; t < 4; ++t) orow[t * 16] = Oacc[t][r] * inv;
  }
}

extern "C" void kernel_launch(void* const* d_in, const int* in_sizes, int n_in,
                              void* d_out, int out_size, void* d_ws, size_t ws_size,
                              hipStream_t stream) {
  const float* x   = (const float*)d_in[0];
  const float* ctx = (const float*)d_in[1];
  const float* Wq  = (const float*)d_in[2];
  const float* Wk  = (const float*)d_in[3];
  const float* Wv  = (const float*)d_in[4];
  const float* Wo  = (const float*)d_in[5];

  uint8_t* wsb = (uint8_t*)d_ws;
  size_t off = 0;
  auto take = [&](size_t bytes) -> void* {
    void* p = wsb + off;
    off = (off + bytes + 255) & ~(size_t)255;
    return p;
  };

  constexpr size_t WBYTES = (size_t)DMODEL * DMODEL;
  constexpr size_t ABYTES = (size_t)TOKENS * DMODEL;
  int8_t* WqQ = (int8_t*)take(WBYTES);
  int8_t* WkQ = (int8_t*)take(WBYTES);
  int8_t* WvQ = (int8_t*)take(WBYTES);
  int8_t* WoQ = (int8_t*)take(WBYTES);
  float*  wScales = (float*)take(4 * sizeof(float));
  float*  xsX = (float*)take(TOKENS * sizeof(float));
  float*  xsC = (float*)take(TOKENS * sizeof(float));
  float*  xsA = (float*)take(TOKENS * sizeof(float));
  int8_t* Xq  = (int8_t*)take(ABYTES);
  int8_t* Cq  = (int8_t*)take(ABYTES);
  int8_t* Aq  = (int8_t*)take(ABYTES);
  _Float16* Qh = (_Float16*)take(ABYTES * 2);
  _Float16* Kh = (_Float16*)take(ABYTES * 2);
  _Float16* Vh = (_Float16*)take(ABYTES * 2);
  float* Att   = (float*)take(ABYTES * 4);

  weight_quant_kernel<<<4, 1024, 0, stream>>>(Wq, Wk, Wv, Wo, WqQ, WkQ, WvQ, WoQ, wScales);

  act_quant_kernel<<<TOKENS, 256, 0, stream>>>(x,   Xq, xsX);
  act_quant_kernel<<<TOKENS, 256, 0, stream>>>(ctx, Cq, xsC);

  const int gemmBlocks = (TOKENS / 32) * (DMODEL / 64) / 8;
  gemm_iu8_kernel<_Float16><<<gemmBlocks, 256, 0, stream>>>(Xq, WqQ, xsX, wScales + 0, Qh, ATTSCALE);
  gemm_iu8_kernel<_Float16><<<gemmBlocks, 256, 0, stream>>>(Cq, WkQ, xsC, wScales + 1, Kh, 1.0f);
  gemm_iu8_kernel<_Float16><<<gemmBlocks, 256, 0, stream>>>(Cq, WvQ, xsC, wScales + 2, Vh, 1.0f);

  attention_kernel<<<BATCH * HEADS * (SEQ / 128), 256, 0, stream>>>(Qh, Kh, Vh, Att);

  act_quant_kernel<<<TOKENS, 256, 0, stream>>>(Att, Aq, xsA);
  gemm_iu8_kernel<float><<<gemmBlocks, 256, 0, stream>>>(Aq, WoQ, xsA, wScales + 3,
                                                         (float*)d_out, 1.0f);

  (void)in_sizes; (void)n_in; (void)out_size; (void)ws_size;
}
